// PointNetSetAbstractionMsg_30391188587131
// MI455X (gfx1250) — matrix operand order repaired and checked by reading, not compiled, not run on hardware
//
#include <hip/hip_runtime.h>
#include <stdint.h>

// ---------------------------------------------------------------------------
// PointNet++ Set Abstraction (MSG) for MI455X / gfx1250
//   - FPS: 1 workgroup per batch, LDS argmax tree
//   - kNN: thread-per-query, register top-k insertion
//   - MLP 1x1 conv: WMMA f32_16x16x32_f16 GEMM
//       * W tile staged to LDS once per block via global_load_async_to_lds_b128
//         (ASYNCcnt / s_wait_asynccnt), A-fragments read with ds_load_b128
//       * X (activations) streamed from global; <=50MB f16, L2-resident
//   - BN: deterministic 2-stage reduction; normalize+ReLU fused with repack
//   - final layer fuses BN+ReLU+max-over-k
// ---------------------------------------------------------------------------

typedef __attribute__((ext_vector_type(16))) _Float16 v16h;
typedef __attribute__((ext_vector_type(8)))  float    v8f;
typedef __attribute__((ext_vector_type(4)))  float    v4f;

#define BB   8
#define NN   8192
#define SS   1024
#define CIN  64
#define EPSV 1e-5f

// ---------------------------------------------------------------------------
// 1) Farthest point sampling: one 1024-thread block per batch (8 pts/thread).
// ---------------------------------------------------------------------------
__global__ void fps_kernel(const float* __restrict__ xyz,   // [B,3,N]
                           int*   __restrict__ fps_idx,     // [B,S]
                           float* __restrict__ nxs,         // [B*S,3]
                           float* __restrict__ out_nx)      // [B,3,S]
{
    const int b   = blockIdx.x;
    const int tid = threadIdx.x;           // 0..1023
    const float* xb = xyz + (size_t)b * 3 * NN;
    const float* yb = xb + NN;
    const float* zb = xb + 2 * NN;

    const int PPT = NN / 1024;             // 8 points per thread (contiguous)
    float px[8], py[8], pz[8], dist[8];
#pragma unroll
    for (int i = 0; i < PPT; ++i) {
        int p = tid * PPT + i;
        px[i] = xb[p]; py[i] = yb[p]; pz[i] = zb[p];
        dist[i] = 1e10f;
    }

    __shared__ float sval[1024];
    __shared__ int   sidx[1024];
    __shared__ float cx, cy, cz;

    int far = 0;
    for (int t = 0; t < SS; ++t) {
        if (tid == 0) {
            float fx = xb[far], fy = yb[far], fz = zb[far];
            fps_idx[b * SS + t] = far;
            nxs[((size_t)b * SS + t) * 3 + 0] = fx;
            nxs[((size_t)b * SS + t) * 3 + 1] = fy;
            nxs[((size_t)b * SS + t) * 3 + 2] = fz;
            out_nx[(size_t)b * 3 * SS + 0 * SS + t] = fx;
            out_nx[(size_t)b * 3 * SS + 1 * SS + t] = fy;
            out_nx[(size_t)b * 3 * SS + 2 * SS + t] = fz;
            cx = fx; cy = fy; cz = fz;
        }
        __syncthreads();

        float qx = cx, qy = cy, qz = cz;
        float best = -1.0f; int bi = tid * PPT;
#pragma unroll
        for (int i = 0; i < PPT; ++i) {
            float dx = px[i] - qx, dy = py[i] - qy, dz = pz[i] - qz;
            float d = dx * dx + dy * dy + dz * dz;
            dist[i] = fminf(dist[i], d);
            if (dist[i] > best) { best = dist[i]; bi = tid * PPT + i; }
        }
        sval[tid] = best; sidx[tid] = bi;
        __syncthreads();
        for (int s = 512; s >= 1; s >>= 1) {
            if (tid < s) {
                float ov = sval[tid + s]; int oi = sidx[tid + s];
                // argmax tie-break: first (lowest) index
                if (ov > sval[tid] || (ov == sval[tid] && oi < sidx[tid])) {
                    sval[tid] = ov; sidx[tid] = oi;
                }
            }
            __syncthreads();
        }
        far = sidx[0];
    }
}

// ---------------------------------------------------------------------------
// 2) kNN: one thread per query; register-resident top-K insertion sort.
// ---------------------------------------------------------------------------
template <int K>
__global__ void knn_kernel(const float* __restrict__ xyz,  // [B,3,N]
                           const float* __restrict__ nxs,  // [B*S,3]
                           int* __restrict__ out)          // [B*S,K]
{
    const int q = blockIdx.x * blockDim.x + threadIdx.x;
    if (q >= BB * SS) return;
    const int b = q / SS;
    const float* xb = xyz + (size_t)b * 3 * NN;
    const float* yb = xb + NN;
    const float* zb = xb + 2 * NN;
    const float qx = nxs[(size_t)q * 3 + 0];
    const float qy = nxs[(size_t)q * 3 + 1];
    const float qz = nxs[(size_t)q * 3 + 2];

    float bd[K]; int bi[K];
#pragma unroll
    for (int j = 0; j < K; ++j) { bd[j] = 3.0e38f; bi[j] = 0; }

    for (int n = 0; n < NN; ++n) {
        float dx = xb[n] - qx, dy = yb[n] - qy, dz = zb[n] - qz;
        float d = dx * dx + dy * dy + dz * dz;
        if (d < bd[K - 1]) {
            float cd = d; int ci = n;
#pragma unroll
            for (int j = 0; j < K; ++j) {
                if (cd < bd[j]) {
                    float td = bd[j]; int ti = bi[j];
                    bd[j] = cd; bi[j] = ci; cd = td; ci = ti;
                }
            }
        }
    }
#pragma unroll
    for (int j = 0; j < K; ++j) out[(size_t)q * K + j] = bi[j];
}

// ---------------------------------------------------------------------------
// 3) Gather + concat -> f16 activation matrix X, column-major [M x 96]
// ---------------------------------------------------------------------------
__global__ void gather_kernel(const float* __restrict__ xyz,
                              const float* __restrict__ points,  // [B,64,N]
                              const float* __restrict__ nxs,     // [B*S,3]
                              const int*   __restrict__ knn,     // [B*S,K]
                              _Float16* __restrict__ X,          // [M x 96]
                              int K, int M)
{
    const int m = blockIdx.x * blockDim.x + threadIdx.x;
    if (m >= M) return;
    const int kk = m % K;
    const int q  = m / K;            // b*S + s
    const int b  = q / SS;
    const int n  = knn[(size_t)q * K + kk];

    const float* xb = xyz + (size_t)b * 3 * NN;
    const float* pb = points + (size_t)b * CIN * NN;
    _Float16* xc = X + (size_t)m * 96;

    xc[0] = (_Float16)(xb[n]          - nxs[(size_t)q * 3 + 0]);
    xc[1] = (_Float16)(xb[NN + n]     - nxs[(size_t)q * 3 + 1]);
    xc[2] = (_Float16)(xb[2 * NN + n] - nxs[(size_t)q * 3 + 2]);
#pragma unroll 8
    for (int c = 0; c < CIN; ++c) xc[3 + c] = (_Float16)pb[(size_t)c * NN + n];
#pragma unroll
    for (int c = 3 + CIN; c < 96; ++c) xc[c] = (_Float16)0.0f;
}

// ---------------------------------------------------------------------------
// 4) Weight fp32 -> f16 with K padding: Wh[Cout x Kpad]
// ---------------------------------------------------------------------------
__global__ void wconv_kernel(const float* __restrict__ w, _Float16* __restrict__ wh,
                             int Cout, int Cin, int Kpad)
{
    int i = blockIdx.x * blockDim.x + threadIdx.x;
    if (i >= Cout * Kpad) return;
    int r = i / Kpad, c = i % Kpad;
    wh[i] = (c < Cin) ? (_Float16)w[(size_t)r * Cin + c] : (_Float16)0.0f;
}

// ---------------------------------------------------------------------------
// 5) WMMA GEMM: Y[Cout x M] = Wh[Cout x KPAD] * X^T  (X is [M x KPAD] f16)
//    8 waves/block, each wave owns 16 columns, keeps all B-fragments in
//    registers, sweeps Cout row tiles.
//    W tile (<=24KB) is staged into LDS once per block with the CDNA5 async
//    global->LDS copy (ASYNCcnt), A-fragments then come from ds_load_b128.
// ---------------------------------------------------------------------------
template <int KPAD>
__global__ void wmma_gemm(const _Float16* __restrict__ Wh,
                          const _Float16* __restrict__ X,
                          const float* __restrict__ bias,
                          float* __restrict__ Y,
                          int Cout, int M)
{
    constexpr int KT = KPAD / 32;
    const int lane = threadIdx.x & 31;
    const int wave = threadIdx.x >> 5;
    const int l    = lane & 15;
    const int half = lane >> 4;
    const int col  = blockIdx.x * 128 + wave * 16 + l;

    __shared__ _Float16 sW[128 * 96];      // max Cout x KPAD tile

    // ---- async stage W tile: global -> LDS, 16B per lane per issue --------
    // All tile sizes (Cout*KPAD*2 in {8192,12288,16384,24576}) are multiples of
    // 4096 = 256 threads * 16B, so the loop trip count is wave-uniform and
    // EXEC stays all-ones (required for the WMMAs below).
    {
        const int bytesTotal = Cout * KPAD * 2;
        for (int o = threadIdx.x * 16; o < bytesTotal; o += 256 * 16) {
            unsigned ldsAddr = (unsigned)(uintptr_t)((char*)sW + o); // addr[31:0] = LDS offset
            const char* gsrc = (const char*)Wh + o;
            asm volatile("global_load_async_to_lds_b128 %0, %1, off"
                         :: "v"(ldsAddr), "v"(gsrc)
                         : "memory");
        }
        asm volatile("s_wait_asynccnt 0x0" ::: "memory");
    }
    __syncthreads();

    union Frag { v16h v; v4f f[2]; };

    // B fragments: lane<16 holds K 0..15 of col, lane>=16 holds K 16..31
    Frag bfrag[KT];
    const _Float16* xc = X + (size_t)col * KPAD + half * 16;
    __builtin_prefetch((const void*)xc, 0, 3);
#pragma unroll
    for (int kt = 0; kt < KT; ++kt) {
        bfrag[kt].f[0] = *(const v4f*)(const void*)(xc + kt * 32);
        bfrag[kt].f[1] = *(const v4f*)(const void*)(xc + kt * 32 + 8);
    }

    for (int rt = 0; rt < Cout; rt += 16) {
        v8f acc = {};
        // A fragment from LDS: lane half selects K 0..15 vs 16..31
        const _Float16* wr = sW + (rt + l) * KPAD + half * 16;
#pragma unroll
        for (int kt = 0; kt < KT; ++kt) {
            Frag a;
            a.f[0] = *(const v4f*)(const void*)(wr + kt * 32);        // ds_load_b128
            a.f[1] = *(const v4f*)(const void*)(wr + kt * 32 + 8);    // ds_load_b128
            acc = __builtin_amdgcn_wmma_f32_16x16x32_f16(
                false, a.v, false, bfrag[kt].v, (short)0, acc, false, false);
        }
        float* yp = Y + (size_t)(rt + half * 8) * M + col;
        const float* bp = bias + rt + half * 8;
#pragma unroll
        for (int r = 0; r < 8; ++r) {
            yp[(size_t)r * M] = acc[r] + bp[r];
        }
    }
}

// ---------------------------------------------------------------------------
// 6) BatchNorm stats: deterministic two-stage sum / sumsq per channel
// ---------------------------------------------------------------------------
__global__ void bnstat1(const float* __restrict__ Y, float* __restrict__ part,
                        int M, int NB)
{
    const int c     = blockIdx.x / NB;
    const int chunk = blockIdx.x % NB;
    const int len   = M / NB;
    const size_t base = (size_t)c * M + (size_t)chunk * len;
    float s = 0.f, s2 = 0.f;
    for (int i = threadIdx.x; i < len; i += 256) {
        float v = Y[base + i];
        s += v; s2 += v * v;
    }
    __shared__ float ss[256], ss2[256];
    ss[threadIdx.x] = s; ss2[threadIdx.x] = s2;
    __syncthreads();
    for (int st = 128; st >= 1; st >>= 1) {
        if (threadIdx.x < st) {
            ss[threadIdx.x]  += ss[threadIdx.x + st];
            ss2[threadIdx.x] += ss2[threadIdx.x + st];
        }
        __syncthreads();
    }
    if (threadIdx.x == 0) {
        part[(size_t)(c * NB + chunk) * 2 + 0] = ss[0];
        part[(size_t)(c * NB + chunk) * 2 + 1] = ss2[0];
    }
}

__global__ void bnstat2(const float* __restrict__ part, float* __restrict__ mv,
                        int NB, int M)
{
    const int c = blockIdx.x;
    float s = 0.f, s2 = 0.f;
    for (int i = threadIdx.x; i < NB; i += 128) {
        s  += part[(size_t)(c * NB + i) * 2 + 0];
        s2 += part[(size_t)(c * NB + i) * 2 + 1];
    }
    __shared__ float ss[128], ss2[128];
    ss[threadIdx.x] = s; ss2[threadIdx.x] = s2;
    __syncthreads();
    for (int st = 64; st >= 1; st >>= 1) {
        if (threadIdx.x < st) {
            ss[threadIdx.x]  += ss[threadIdx.x + st];
            ss2[threadIdx.x] += ss2[threadIdx.x + st];
        }
        __syncthreads();
    }
    if (threadIdx.x == 0) {
        float mean = ss[0] / (float)M;
        mv[c * 2 + 0] = mean;
        mv[c * 2 + 1] = ss2[0] / (float)M - mean * mean;   // biased var (ddof=0)
    }
}

// ---------------------------------------------------------------------------
// 7) BN + ReLU + repack to next layer's f16 col-major X [M x Cout]
// ---------------------------------------------------------------------------
__global__ void bnrelu_kernel(const float* __restrict__ Y, const float* __restrict__ mv,
                              const float* __restrict__ g, const float* __restrict__ be,
                              _Float16* __restrict__ Xn, int Cout, int M)
{
    const size_t i = (size_t)blockIdx.x * blockDim.x + threadIdx.x;
    if (i >= (size_t)Cout * M) return;
    const int c = (int)(i / M);
    const int m = (int)(i % M);
    float mean = mv[c * 2], var = mv[c * 2 + 1];
    float v = (Y[i] - mean) * rsqrtf(var + EPSV) * g[c] + be[c];
    Xn[(size_t)m * Cout + c] = (_Float16)fmaxf(v, 0.0f);
}

// ---------------------------------------------------------------------------
// 8) Final layer: BN + ReLU + max over K samples -> features [B,256,S]
// ---------------------------------------------------------------------------
__global__ void bnrelu_max_kernel(const float* __restrict__ Y, const float* __restrict__ mv,
                                  const float* __restrict__ g, const float* __restrict__ be,
                                  float* __restrict__ feat, int M, int K, int chanOff)
{
    const int i = blockIdx.x * blockDim.x + threadIdx.x;
    if (i >= BB * 128 * SS) return;
    const int b = i / (128 * SS);
    const int c = (i / SS) % 128;
    const int s = i % SS;
    float mean = mv[c * 2], var = mv[c * 2 + 1];
    float sc = rsqrtf(var + EPSV) * g[c];
    float sh = be[c] - mean * sc;
    const size_t base = (size_t)c * M + (size_t)(b * SS + s) * K;
    float best = -3.0e38f;
    for (int kk = 0; kk < K; ++kk) best = fmaxf(best, Y[base + kk] * sc + sh);
    feat[(size_t)b * 256 * SS + (size_t)(chanOff + c) * SS + s] = fmaxf(best, 0.0f);
}

// ---------------------------------------------------------------------------
// Host orchestration
// ---------------------------------------------------------------------------
extern "C" void kernel_launch(void* const* d_in, const int* in_sizes, int n_in,
                              void* d_out, int out_size, void* d_ws, size_t ws_size,
                              hipStream_t stream)
{
    (void)in_sizes; (void)n_in; (void)out_size; (void)ws_size;
    const float* xyz    = (const float*)d_in[0];
    const float* points = (const float*)d_in[1];

    char* ws = (char*)d_ws;
    size_t off = 0;
    auto alloc = [&](size_t bytes) -> void* {
        void* p = ws + off;
        off = (off + bytes + 255) & ~(size_t)255;
        return p;
    };

    const size_t Mmax = (size_t)BB * SS * 32;
    int*      fps_idx = (int*)     alloc((size_t)BB * SS * sizeof(int));
    float*    nxs     = (float*)   alloc((size_t)BB * SS * 3 * sizeof(float));
    int*      knn     = (int*)     alloc((size_t)BB * SS * 32 * sizeof(int));
    _Float16* Wh      = (_Float16*)alloc((size_t)128 * 96 * sizeof(_Float16));
    _Float16* Xa      = (_Float16*)alloc(Mmax * 96 * sizeof(_Float16));
    _Float16* Xb      = (_Float16*)alloc(Mmax * 96 * sizeof(_Float16));
    float*    Yb      = (float*)   alloc((size_t)128 * Mmax * sizeof(float));
    const int NB = 128;
    float*    part    = (float*)   alloc((size_t)128 * NB * 2 * sizeof(float));
    float*    mv      = (float*)   alloc((size_t)128 * 2 * sizeof(float));

    float* out_nx = (float*)d_out;                 // [B,3,S]
    float* feat   = out_nx + (size_t)BB * 3 * SS;  // [B,256,S]

    // 1) FPS
    fps_kernel<<<BB, 1024, 0, stream>>>(xyz, fps_idx, nxs, out_nx);

    const int mlps[2][3] = {{64, 64, 128}, {64, 96, 128}};
    for (int i = 0; i < 2; ++i) {
        const int K = (i == 0) ? 16 : 32;
        const int M = BB * SS * K;           // 131072 / 262144 (div by 128)

        // 2) kNN
        if (i == 0) knn_kernel<16><<<(BB * SS + 255) / 256, 256, 0, stream>>>(xyz, nxs, knn);
        else        knn_kernel<32><<<(BB * SS + 255) / 256, 256, 0, stream>>>(xyz, nxs, knn);

        // 3) gather + concat -> f16 X [M x 96]
        gather_kernel<<<(M + 255) / 256, 256, 0, stream>>>(xyz, points, nxs, knn, Xa, K, M);

        _Float16* Xcur = Xa;
        _Float16* Xnxt = Xb;
        int Kin = 3 + CIN;                   // 67
        for (int j = 0; j < 3; ++j) {
            const int Cout = mlps[i][j];
            const int Kpad = (Kin + 31) & ~31;
            const float* w   = (const float*)d_in[2 + (i * 3 + j) * 4 + 0];
            const float* bi_ = (const float*)d_in[2 + (i * 3 + j) * 4 + 1];
            const float* g   = (const float*)d_in[2 + (i * 3 + j) * 4 + 2];
            const float* be  = (const float*)d_in[2 + (i * 3 + j) * 4 + 3];

            // 4) weights -> f16 padded
            wconv_kernel<<<(Cout * Kpad + 255) / 256, 256, 0, stream>>>(w, Wh, Cout, Kin, Kpad);

            // 5) WMMA GEMM (async-LDS staged W)
            if (Kpad == 96) wmma_gemm<96><<<dim3(M / 128), 256, 0, stream>>>(Wh, Xcur, bi_, Yb, Cout, M);
            else            wmma_gemm<64><<<dim3(M / 128), 256, 0, stream>>>(Wh, Xcur, bi_, Yb, Cout, M);

            // 6) BN stats
            bnstat1<<<Cout * NB, 256, 0, stream>>>(Yb, part, M, NB);
            bnstat2<<<Cout, 128, 0, stream>>>(part, mv, NB, M);

            if (j < 2) {
                // 7) BN + ReLU -> next X
                unsigned int nblk = (unsigned int)(((size_t)Cout * M + 255) / 256);
                bnrelu_kernel<<<nblk, 256, 0, stream>>>(Yb, mv, g, be, Xnxt, Cout, M);
                _Float16* t = Xcur; Xcur = Xnxt; Xnxt = t;
            } else {
                // 8) BN + ReLU + max over K -> output features
                bnrelu_max_kernel<<<(BB * 128 * SS + 255) / 256, 256, 0, stream>>>(
                    Yb, mv, g, be, feat, M, K, i * 128);
            }
            Kin = Cout;
        }
    }
}
